// MS_SSA_Conv_STAtten_79216376807545
// MI455X (gfx1250) — matrix operand order repaired and checked by reading, not compiled, not run on hardware
//
#include <hip/hip_runtime.h>

// MS-SSA spiking attention block for MI455X (gfx1250, wave32, WMMA).
// All matmuls use v_wmma_f32_16x16x32_f16: spike activations are binary {0,1}
// (exact in f16), weights pre-converted f32->f16, f32 accumulation.
// Big GEMMs use 32(M)x64(N) wave tiles: 8 WMMAs per 6 fragment loads per
// k-step, halving B-operand L2 traffic vs a 16x64 tile.

typedef __attribute__((ext_vector_type(16))) _Float16 v16h;
typedef __attribute__((ext_vector_type(8)))  _Float16 v8h;
typedef __attribute__((ext_vector_type(8)))  float    v8f;

namespace {
constexpr int Tn = 8, Bn = 16, Cn = 256, Vn = 1024, Hn = 8, Dn = 32;
constexpr long long En = (long long)Tn * Bn * Cn * Vn;   // 33,554,432
}

union U16h { v16h v; v8h h[2]; };
union U8f  { v8f v;  float f[8]; };

// ---- WMMA fragment loaders -------------------------------------------------
// A operand: 16x32 (MxK) f16, memory [M][K] row-major, K-contiguous.
__device__ __forceinline__ v16h frag_a(const _Float16* p, int ld, int lane) {
  int r  = lane & 15;
  int kb = (lane >> 4) << 3;           // 0 or 8
  const _Float16* q = p + (long long)r * ld + kb;
  U16h u;
  u.h[0] = *(const v8h*)(q);           // K = kb .. kb+7
  u.h[1] = *(const v8h*)(q + 16);      // K = 16+kb .. 16+kb+7
  return u.v;
}

// B operand: 32x16 (KxN) f16, memory [N][K] row-major (N-major, K-contiguous).
__device__ __forceinline__ v16h frag_b(const _Float16* p, int ld, int lane) {
  int n  = lane & 15;
  int kb = (lane >> 4) << 3;           // 0 or 8
  const _Float16* q = p + (long long)n * ld + kb;
  U16h u;
  u.h[0] = *(const v8h*)(q);
  u.h[1] = *(const v8h*)(q + 16);
  return u.v;
}

__device__ __forceinline__ v8f wmma_f16(v16h a, v16h b, v8f c) {
  return __builtin_amdgcn_wmma_f32_16x16x32_f16(false, a, false, b,
                                                (short)0, c, false, false);
}

// ---- small helper kernels --------------------------------------------------
__global__ void k_w2h(const float* __restrict__ in, _Float16* __restrict__ out, int n) {
  int i = blockIdx.x * 256 + threadIdx.x;
  if (i < n) out[i] = (_Float16)in[i];
}

// Fold proj bias into BN affine: r = P*sC[o] + bC[o]  (then +identity).
__global__ void k_bnproj(const float* __restrict__ pb, const float* __restrict__ bnp,
                         float* __restrict__ sC, float* __restrict__ bC) {
  int c = threadIdx.x;
  float g = bnp[c], be = bnp[256 + c], m = bnp[512 + c], va = bnp[768 + c];
  float s = g * rsqrtf(va + 1e-5f);
  sC[c] = s;
  bC[c] = (pb[c] - m) * s + be;
}

// ---- shortcut LIF: x[T,B,C,V] f32 -> xs_t[T,B,V,C] f16 (LDS-tiled transpose)
__global__ void k_lif_x(const float* __restrict__ x, _Float16* __restrict__ xs_t) {
  __shared__ _Float16 tile[32][33];
  int bi = blockIdx.x;
  int vt = bi & 31;            // 32 v-tiles
  int ct = (bi >> 5) & 7;      // 8 c-tiles
  int b  = bi >> 8;            // 16 batches
  int tid = threadIdx.x;
  int vl = tid & 31;
  int c0 = tid >> 5;           // 0..7
  float vm[4] = {0.f, 0.f, 0.f, 0.f};
  for (int t = 0; t < Tn; ++t) {
    long long base = (((long long)(t * Bn + b) * Cn + ct * 32) * Vn) + vt * 32;
#pragma unroll
    for (int j = 0; j < 4; ++j) {
      int cl = c0 + 8 * j;
      float xv = x[base + (long long)cl * Vn + vl];
      vm[j] = vm[j] + (xv - vm[j]) * 0.5f;            // tau = 2
      float s = (vm[j] >= 1.0f) ? 1.0f : 0.0f;        // v_th = 1
      vm[j] *= (1.0f - s);                            // hard reset
      tile[vl][cl] = (_Float16)s;
    }
    __syncthreads();
    long long obase = (((long long)(t * Bn + b) * Vn + vt * 32) * Cn) + ct * 32;
#pragma unroll
    for (int j = 0; j < 4; ++j) {
      int r = c0 + 8 * j;
      xs_t[obase + (long long)r * Cn + vl] = tile[r][vl];
    }
    __syncthreads();
  }
}

// ---- branch GEMM: Y[sl][o][v] = sum_c W[o][c] * Xt[sl][v][c]  (f16 out) ----
// wave tile: 32(M) x 64(N), K=256 in 8 WMMA steps; 8 WMMAs / 6 frag loads.
__global__ void k_gemm_branch(const _Float16* __restrict__ W,
                              const _Float16* __restrict__ Xt,
                              _Float16* __restrict__ Y) {
  int lane = threadIdx.x & 31;
  int w  = blockIdx.x * 8 + (threadIdx.x >> 5);
  int vg = w & 15;             // 16 groups of 64 v
  int og = (w >> 4) & 7;       // 8 groups of 32 o
  int sl = w >> 7;             // 128 (t,b) slices
  const _Float16* Arow  = W + (long long)og * 32 * Cn;
  const _Float16* Bbase = Xt + (long long)sl * Vn * Cn + (long long)vg * 64 * Cn;
  v8f acc[8] = {};             // [ti=o-subtile][j=v-subtile]
  for (int kk = 0; kk < Cn; kk += 32) {
    __builtin_prefetch(Bbase + kk + 32, 0, 1);
    v16h af0 = frag_a(Arow + kk, Cn, lane);
    v16h af1 = frag_a(Arow + 16 * Cn + kk, Cn, lane);
    v16h b0 = frag_b(Bbase + 0 * 16 * Cn + kk, Cn, lane);
    v16h b1 = frag_b(Bbase + 1 * 16 * Cn + kk, Cn, lane);
    v16h b2 = frag_b(Bbase + 2 * 16 * Cn + kk, Cn, lane);
    v16h b3 = frag_b(Bbase + 3 * 16 * Cn + kk, Cn, lane);
    acc[0] = wmma_f16(af0, b0, acc[0]);
    acc[1] = wmma_f16(af0, b1, acc[1]);
    acc[2] = wmma_f16(af0, b2, acc[2]);
    acc[3] = wmma_f16(af0, b3, acc[3]);
    acc[4] = wmma_f16(af1, b0, acc[4]);
    acc[5] = wmma_f16(af1, b1, acc[5]);
    acc[6] = wmma_f16(af1, b2, acc[6]);
    acc[7] = wmma_f16(af1, b3, acc[7]);
  }
  int n  = lane & 15;
  int mb = (lane >> 4) * 8;
  long long ob = (long long)sl * Cn * Vn + (long long)(og * 32) * Vn + vg * 64;
#pragma unroll
  for (int ti = 0; ti < 2; ++ti) {
#pragma unroll
    for (int j = 0; j < 4; ++j) {
      U8f u; u.v = acc[ti * 4 + j];
#pragma unroll
      for (int g = 0; g < 8; ++g) {
        Y[ob + (long long)(ti * 16 + mb + g) * Vn + j * 16 + n] = (_Float16)u.f[g];
      }
    }
  }
}

// ---- BN (optional) + LIF scan over T --------------------------------------
__global__ void k_lif_scan(const _Float16* __restrict__ Y, const float* __restrict__ bnp,
                           float vth, _Float16* __restrict__ S, int transposed,
                           float* __restrict__ vout) {
  long long idx = (long long)blockIdx.x * 256 + threadIdx.x;  // B*C*V threads
  int b = (int)(idx >> 18);
  int c = (int)((idx >> 10) & 255);
  int v = (int)(idx & 1023);
  float sc = 1.0f, sh = 0.0f;
  if (bnp) {
    float g = bnp[c], be = bnp[256 + c], m = bnp[512 + c], va = bnp[768 + c];
    sc = g * rsqrtf(va + 1e-5f);
    sh = be - m * sc;
  }
  float vm = 0.0f;
  for (int t = 0; t < Tn; ++t) {
    long long ii = ((long long)(t * Bn + b) * Cn + c) * Vn + v;
    float yv = (float)Y[ii] * sc + sh;
    vm += (yv - vm) * 0.5f;
    float s = (vm >= vth) ? 1.0f : 0.0f;
    vm *= (1.0f - s);
    long long oo = transposed ? (((long long)(t * Bn + b) * Vn + v) * Cn + c) : ii;
    S[oo] = (_Float16)s;
    if (vout) {
      int h = c >> 5, d = c & 31;
      vout[(((long long)(t * Bn + b) * Hn + h) * Vn + v) * Dn + d] = s;
    }
  }
}

// ---- attention stage 1: attnT[e][d] = (1/N) * sum_l V[l,e] K[l,d] ----------
__global__ void k_attn(const _Float16* __restrict__ SV, const _Float16* __restrict__ SK,
                       _Float16* __restrict__ AT) {
  int lane = threadIdx.x & 31;
  int w  = blockIdx.x * 8 + (threadIdx.x >> 5);   // 512 waves
  int h  = w & 7;
  int b  = (w >> 3) & 15;
  int nc = w >> 7;
  v8f c00 = {}, c01 = {}, c10 = {}, c11 = {};     // [etile][dtile]
  for (int ct = 0; ct < 2; ++ct) {
    int t = nc * 2 + ct;
    const _Float16* vb = SV + ((long long)(t * Bn + b) * Cn + h * Dn) * Vn;
    const _Float16* kb = SK + ((long long)(t * Bn + b) * Cn + h * Dn) * Vn;
    for (int kk = 0; kk < Vn; kk += 32) {
      v16h a0 = frag_a(vb + kk, Vn, lane);
      v16h a1 = frag_a(vb + 16 * Vn + kk, Vn, lane);
      v16h b0 = frag_b(kb + kk, Vn, lane);
      v16h b1 = frag_b(kb + 16 * Vn + kk, Vn, lane);
      c00 = wmma_f16(a0, b0, c00);
      c01 = wmma_f16(a0, b1, c01);
      c10 = wmma_f16(a1, b0, c10);
      c11 = wmma_f16(a1, b1, c11);
    }
  }
  long long base = ((long long)(nc * Bn + b) * Hn + h) * Dn * Dn;
  int n = lane & 15, mb = (lane >> 4) * 8;
  const float inv = 1.0f / (float)Vn;
  U8f u00, u01, u10, u11; u00.v = c00; u01.v = c01; u10.v = c10; u11.v = c11;
#pragma unroll
  for (int g = 0; g < 8; ++g) {
    AT[base + (long long)( 0 + mb + g) * Dn +  0 + n] = (_Float16)(u00.f[g] * inv);
    AT[base + (long long)( 0 + mb + g) * Dn + 16 + n] = (_Float16)(u01.f[g] * inv);
    AT[base + (long long)(16 + mb + g) * Dn +  0 + n] = (_Float16)(u10.f[g] * inv);
    AT[base + (long long)(16 + mb + g) * Dn + 16 + n] = (_Float16)(u11.f[g] * inv);
  }
}

// ---- attention stage 2: R[e][l] = sum_d attnT[e][d] * q[l][d]; K=32 one step
__global__ void k_outgemm(const _Float16* __restrict__ AT, const _Float16* __restrict__ SQT,
                          _Float16* __restrict__ AOUT) {
  int lane = threadIdx.x & 31;
  int w  = blockIdx.x * 8 + (threadIdx.x >> 5);   // 16384 waves
  int lg = w & 31;             // group of 4 l-tiles
  int h  = (w >> 5) & 7;
  int b  = (w >> 8) & 15;
  int nc = w >> 12;
  long long atb = ((long long)(nc * Bn + b) * Hn + h) * Dn * Dn;
  v16h a0 = frag_a(AT + atb, Dn, lane);               // e 0..15, d 0..31
  v16h a1 = frag_a(AT + atb + 16 * Dn, Dn, lane);     // e 16..31
  int n = lane & 15, mb = (lane >> 4) * 8;
#pragma unroll
  for (int j = 0; j < 4; ++j) {
    int lt = lg * 4 + j;       // 0..127 ; l = lt*16+n ; ct = lt/64
    int ct = lt >> 6;
    int nt = lt & 63;
    int t  = nc * 2 + ct;
    const _Float16* qb = SQT + ((long long)(t * Bn + b) * Vn + nt * 16) * Cn + h * Dn;
    v16h bf = frag_b(qb, Cn, lane);
    v8f c0 = {}, c1 = {};
    c0 = wmma_f16(a0, bf, c0);
    c1 = wmma_f16(a1, bf, c1);
    U8f u0, u1; u0.v = c0; u1.v = c1;
    long long ob = ((long long)(t * Bn + b) * Cn + h * Dn) * Vn + nt * 16;
#pragma unroll
    for (int g = 0; g < 8; ++g) {
      AOUT[ob + (long long)( 0 + mb + g) * Vn + n] = (_Float16)u0.f[g];
      AOUT[ob + (long long)(16 + mb + g) * Vn + n] = (_Float16)u1.f[g];
    }
  }
}

// ---- projection GEMM + fused BN affine + residual (f32 out) ----------------
// Same 32x64 wave tile as the branch GEMM.
__global__ void k_proj(const _Float16* __restrict__ W, const _Float16* __restrict__ Xt,
                       const float* __restrict__ sC, const float* __restrict__ bC,
                       const float* __restrict__ xin, float* __restrict__ out) {
  int lane = threadIdx.x & 31;
  int w  = blockIdx.x * 8 + (threadIdx.x >> 5);
  int vg = w & 15;
  int og = (w >> 4) & 7;
  int sl = w >> 7;
  const _Float16* Arow  = W + (long long)og * 32 * Cn;
  const _Float16* Bbase = Xt + (long long)sl * Vn * Cn + (long long)vg * 64 * Cn;
  v8f acc[8] = {};
  for (int kk = 0; kk < Cn; kk += 32) {
    __builtin_prefetch(Bbase + kk + 32, 0, 1);
    v16h af0 = frag_a(Arow + kk, Cn, lane);
    v16h af1 = frag_a(Arow + 16 * Cn + kk, Cn, lane);
    v16h b0 = frag_b(Bbase + 0 * 16 * Cn + kk, Cn, lane);
    v16h b1 = frag_b(Bbase + 1 * 16 * Cn + kk, Cn, lane);
    v16h b2 = frag_b(Bbase + 2 * 16 * Cn + kk, Cn, lane);
    v16h b3 = frag_b(Bbase + 3 * 16 * Cn + kk, Cn, lane);
    acc[0] = wmma_f16(af0, b0, acc[0]);
    acc[1] = wmma_f16(af0, b1, acc[1]);
    acc[2] = wmma_f16(af0, b2, acc[2]);
    acc[3] = wmma_f16(af0, b3, acc[3]);
    acc[4] = wmma_f16(af1, b0, acc[4]);
    acc[5] = wmma_f16(af1, b1, acc[5]);
    acc[6] = wmma_f16(af1, b2, acc[6]);
    acc[7] = wmma_f16(af1, b3, acc[7]);
  }
  int n  = lane & 15;
  int mb = (lane >> 4) * 8;
  long long ob = (long long)sl * Cn * Vn + (long long)(og * 32) * Vn + vg * 64;
#pragma unroll
  for (int ti = 0; ti < 2; ++ti) {
#pragma unroll
    for (int j = 0; j < 4; ++j) {
      U8f u; u.v = acc[ti * 4 + j];
#pragma unroll
      for (int g = 0; g < 8; ++g) {
        int o = og * 32 + ti * 16 + mb + g;
        long long ii = ob + (long long)(ti * 16 + mb + g) * Vn + j * 16 + n;
        out[ii] = u.f[g] * sC[o] + bC[o] + xin[ii];
      }
    }
  }
}

// ---- host side -------------------------------------------------------------
extern "C" void kernel_launch(void* const* d_in, const int* in_sizes, int n_in,
                              void* d_out, int out_size, void* d_ws, size_t ws_size,
                              hipStream_t stream) {
  (void)in_sizes; (void)n_in; (void)out_size; (void)ws_size;
  const float* x   = (const float*)d_in[0];
  const float* qw  = (const float*)d_in[1];
  const float* kw  = (const float*)d_in[2];
  const float* vw  = (const float*)d_in[3];
  const float* pw  = (const float*)d_in[4];
  const float* pb  = (const float*)d_in[5];
  const float* bnq = (const float*)d_in[6];
  const float* bnk = (const float*)d_in[7];
  const float* bnv = (const float*)d_in[8];
  const float* bnp = (const float*)d_in[9];

  char* ws = (char*)d_ws;
  _Float16* wq16 = (_Float16*)(ws);
  _Float16* wk16 = wq16 + Cn * Cn;
  _Float16* wv16 = wk16 + Cn * Cn;
  _Float16* wp16 = wv16 + Cn * Cn;
  float* sC = (float*)(ws + 524288);
  float* bC = sC + Cn;
  const long long REG = En * 2;  // bytes per big f16 region
  _Float16* xs_t = (_Float16*)(ws + (1 << 20));
  _Float16* yq   = (_Float16*)((char*)xs_t + 1 * REG);
  _Float16* yk   = (_Float16*)((char*)xs_t + 2 * REG);
  _Float16* yv   = (_Float16*)((char*)xs_t + 3 * REG);
  _Float16* sqt  = (_Float16*)((char*)xs_t + 4 * REG);
  _Float16* sk   = (_Float16*)((char*)xs_t + 5 * REG);
  _Float16* sv   = (_Float16*)((char*)xs_t + 6 * REG);
  _Float16* atT  = (_Float16*)((char*)xs_t + 7 * REG);
  _Float16* aout = xs_t;   // reuse: xs_t dead after branch GEMMs
  _Float16* spt  = yq;     // reuse: yq dead after its BN+LIF

  float* out1 = (float*)d_out;        // y + identity  [T,B,C,V]
  float* out2 = out1 + En;            // v spikes      [T,B,H,N,D]

  k_w2h<<<256, 256, 0, stream>>>(qw, wq16, Cn * Cn);
  k_w2h<<<256, 256, 0, stream>>>(kw, wk16, Cn * Cn);
  k_w2h<<<256, 256, 0, stream>>>(vw, wv16, Cn * Cn);
  k_w2h<<<256, 256, 0, stream>>>(pw, wp16, Cn * Cn);
  k_bnproj<<<1, 256, 0, stream>>>(pb, bnp, sC, bC);

  k_lif_x<<<4096, 256, 0, stream>>>(x, xs_t);

  k_gemm_branch<<<2048, 256, 0, stream>>>(wq16, xs_t, yq);
  k_gemm_branch<<<2048, 256, 0, stream>>>(wk16, xs_t, yk);
  k_gemm_branch<<<2048, 256, 0, stream>>>(wv16, xs_t, yv);

  k_lif_scan<<<16384, 256, 0, stream>>>(yq, bnq, 1.0f, sqt, 1, nullptr);
  k_lif_scan<<<16384, 256, 0, stream>>>(yk, bnk, 1.0f, sk, 0, nullptr);
  k_lif_scan<<<16384, 256, 0, stream>>>(yv, bnv, 1.0f, sv, 0, out2);

  k_attn<<<64, 256, 0, stream>>>(sv, sk, atT);
  k_outgemm<<<2048, 256, 0, stream>>>(atT, sqt, aout);

  k_lif_scan<<<16384, 256, 0, stream>>>(aout, nullptr, 0.5f, spt, 1, nullptr);

  k_proj<<<2048, 256, 0, stream>>>(wp16, spt, sC, bC, x, out1);
}
